// GNN_16527034155612
// MI455X (gfx1250) — compile-verified, matrix operand order checked by reading, not run on hardware
//
#include <hip/hip_runtime.h>

#define NN   100000   // nodes
#define NG   16       // graphs
#define NPG  6250     // nodes per graph
#define NE   640000   // edges
#define F    128      // feature dim (in == hidden)
#define OUTC 20
#define BN_EPS 1e-5f

typedef __attribute__((ext_vector_type(16))) _Float16 v16h;
typedef __attribute__((ext_vector_type(8)))  float    v8f;

// ---------------- WMMA fragment helpers (ISA 7.12.2 layouts) ----------------
// 16-bit A 16x32: lanes 0-15 hold M=lane, K kb..kb+7 & kb+16..kb+23 with
// kb = k0 (+8 for lanes 16-31). Element e -> K = kb+e (e<8), kb+8+e (e>=8).
__device__ __forceinline__ int frag_k(int e, int kb) {
    return (e < 8) ? (kb + e) : (kb + 8 + e);
}

// A fragment from f32 row-major: two contiguous 32B runs per lane -> float4 x4
__device__ __forceinline__ v16h load_fragA_f32(const float* __restrict__ A, int ld,
                                               int row0, int k0, int lane, int nrows) {
    int row = row0 + (lane & 15);
    if (row >= nrows) row = nrows - 1;   // duplicate-row clamp; stores are guarded
    int kb = k0 + ((lane >> 4) << 3);
    const float4* p = (const float4*)(A + (long)row * ld + kb);  // 32B aligned
    float4 a0 = p[0];   // K = kb..kb+3
    float4 a1 = p[1];   // K = kb+4..kb+7
    float4 a2 = p[4];   // K = kb+16..kb+19
    float4 a3 = p[5];   // K = kb+20..kb+23
    v16h f;
    f[0]=(_Float16)a0.x;  f[1]=(_Float16)a0.y;  f[2]=(_Float16)a0.z;  f[3]=(_Float16)a0.w;
    f[4]=(_Float16)a1.x;  f[5]=(_Float16)a1.y;  f[6]=(_Float16)a1.z;  f[7]=(_Float16)a1.w;
    f[8]=(_Float16)a2.x;  f[9]=(_Float16)a2.y;  f[10]=(_Float16)a2.z; f[11]=(_Float16)a2.w;
    f[12]=(_Float16)a3.x; f[13]=(_Float16)a3.y; f[14]=(_Float16)a3.z; f[15]=(_Float16)a3.w;
    return f;
}

// B fragment from pre-packed fragment-major buffer: one contiguous v16h per lane
// (32B/lane, 1KB contiguous per wave -> two coalesced global_load_b128)
__device__ __forceinline__ v16h load_fragB_packed(const _Float16* __restrict__ Wp,
                                                  int frag, int lane) {
    return ((const v16h*)Wp)[frag * 32 + lane];
}

// ---------------- utility kernels ----------------
__global__ void zero_kernel(float* __restrict__ p, long n) {
    long i = (long)blockIdx.x * blockDim.x + threadIdx.x;
    long st = (long)gridDim.x * blockDim.x;
    for (; i < n; i += st) p[i] = 0.0f;
}

// Pack a 128x128 f32 row-major weight into f16 WMMA B-fragment-major layout.
// frag f = kt*8 + nt (kt = K-tile/32, nt = N-tile/16);
// Wp[f*512 + lane*16 + e] = W[ frag_k(e, kt*32 + (lane>>4)*8) * 128 + nt*16 + (lane&15) ]
__global__ __launch_bounds__(256)
void pack_w_kernel(const float* __restrict__ W, _Float16* __restrict__ Wp) {
    int t = blockIdx.x * 256 + threadIdx.x;   // 0..2047 = 64 frags * 32 lanes
    int f = t >> 5, lane = t & 31;
    int kt = f >> 3, nt = f & 7;
    int kb = kt * 32 + ((lane >> 4) << 3);
    int n  = nt * 16 + (lane & 15);
    _Float16* d = Wp + f * 512 + lane * 16;
#pragma unroll
    for (int e = 0; e < 16; ++e) d[e] = (_Float16)W[frag_k(e, kb) * F + n];
}

// per-graph column sums: one block per graph, thread = column (coalesced rows)
__global__ __launch_bounds__(128)
void vsum_kernel(const float* __restrict__ X, float* __restrict__ vsum) {
    int g = blockIdx.x, j = threadIdx.x;
    const float* p = X + (long)g * NPG * F + j;
    float s = 0.0f;
    for (int r = 0; r < NPG; ++r) s += p[(long)r * F];
    vsum[g * F + j] = s;
}

// vx = relu(vsum @ Wvn + bvn): M=16, N=128, K=128 -> 1 block, 8 waves, wave = N-tile
__global__ __launch_bounds__(256)
void vn_mlp_kernel(const float* __restrict__ vsum, const _Float16* __restrict__ Wp,
                   const float* __restrict__ bvn, float* __restrict__ vx) {
    int lane = threadIdx.x & 31;
    int nt   = threadIdx.x >> 5;     // 0..7 N-tile
    v8f c = {};
#pragma unroll
    for (int kt = 0; kt < 4; ++kt) {
        v16h a = load_fragA_f32(vsum, F, 0, kt * 32, lane, NG);
        v16h b = load_fragB_packed(Wp, kt * 8 + nt, lane);
        c = __builtin_amdgcn_wmma_f32_16x16x32_f16(false, a, false, b, (short)0, c, false, false);
    }
#pragma unroll
    for (int r = 0; r < 8; ++r) {
        int row = r + ((lane >> 4) << 3);
        int col = nt * 16 + (lane & 15);
        float v = c[r] + bvn[col];
        vx[row * F + col] = v > 0.0f ? v : 0.0f;
    }
}

// xb = xin + vx[node/NPG]
__global__ void bcast_add_kernel(const float* __restrict__ Xin, const float* __restrict__ vx,
                                 float* __restrict__ Xb) {
    long i = (long)blockIdx.x * blockDim.x + threadIdx.x;
    long total = (long)NN * F;
    long st = (long)gridDim.x * blockDim.x;
    for (; i < total; i += st) {
        int node = (int)(i / F), j = (int)(i % F);
        Xb[i] = Xin[i] + vx[(node / NPG) * F + j];
    }
}

// agg[dst] += x[src]: one wave per edge, lane carries 4 floats (float4 load)
__global__ __launch_bounds__(256)
void scatter_add_kernel(const float* __restrict__ X, const long long* __restrict__ ei,
                        float* __restrict__ agg) {
    int gw = (int)(((long)blockIdx.x * blockDim.x + threadIdx.x) >> 5);
    int lane = threadIdx.x & 31;
    if (gw >= NE) return;
    long s = (long)ei[gw];
    long d = (long)ei[NE + gw];
    float4 v = ((const float4*)(X + s * F))[lane];
    float* ad = agg + d * F + lane * 4;
    atomicAdd(ad + 0, v.x); atomicAdd(ad + 1, v.y);
    atomicAdd(ad + 2, v.z); atomicAdd(ad + 3, v.w);
}

// Y = A0@W0 + A1@W1 + bias : 128-row stripe per block, wave = 16-row band x full N=128
__global__ __launch_bounds__(256)
void gemm_dual_kernel(const float* __restrict__ A0, const float* __restrict__ A1,
                      const _Float16* __restrict__ W0p, const _Float16* __restrict__ W1p,
                      const float* __restrict__ bias, float* __restrict__ Y, int nrows) {
    int lane = threadIdx.x & 31;
    int wave = threadIdx.x >> 5;
    int row0 = blockIdx.x * 128 + wave * 16;
    if (row0 >= nrows) return;          // wave-uniform, EXEC stays full for WMMA
    v8f zero = {};
    v8f c[8];
#pragma unroll
    for (int t = 0; t < 8; ++t) c[t] = zero;

    // prefetch the A1 stripe this wave will read in phase 2 (global_prefetch_b8)
    {
        int rowp = row0 + (lane & 15);
        if (rowp >= nrows) rowp = nrows - 1;
        const char* p1 = (const char*)(A1 + (long)rowp * F);
        __builtin_prefetch(p1 + (lane >> 4) * 256, 0, 1);
        __builtin_prefetch(p1 + (lane >> 4) * 256 + 128, 0, 1);
    }

#pragma unroll
    for (int kt = 0; kt < 4; ++kt) {
        v16h a = load_fragA_f32(A0, F, row0, kt * 32, lane, nrows);
#pragma unroll
        for (int nt = 0; nt < 8; ++nt) {
            v16h b = load_fragB_packed(W0p, kt * 8 + nt, lane);
            c[nt] = __builtin_amdgcn_wmma_f32_16x16x32_f16(false, a, false, b, (short)0, c[nt], false, false);
        }
    }
#pragma unroll
    for (int kt = 0; kt < 4; ++kt) {
        v16h a = load_fragA_f32(A1, F, row0, kt * 32, lane, nrows);
#pragma unroll
        for (int nt = 0; nt < 8; ++nt) {
            v16h b = load_fragB_packed(W1p, kt * 8 + nt, lane);
            c[nt] = __builtin_amdgcn_wmma_f32_16x16x32_f16(false, a, false, b, (short)0, c[nt], false, false);
        }
    }
#pragma unroll
    for (int nt = 0; nt < 8; ++nt) {
        int col = nt * 16 + (lane & 15);
        float bv = bias[col];
#pragma unroll
        for (int r = 0; r < 8; ++r) {
            int row = row0 + r + ((lane >> 4) << 3);
            if (row < nrows) Y[(long)row * F + col] = c[nt][r] + bv;
        }
    }
}

// column sum / sum-of-squares, one local accumulation per block then atomics
__global__ __launch_bounds__(128)
void colstats_kernel(const float* __restrict__ Y, float* __restrict__ csum,
                     float* __restrict__ csq) {
    int j = threadIdx.x;
    float s = 0.0f, s2 = 0.0f;
    for (int r = blockIdx.x; r < NN; r += gridDim.x) {
        float v = Y[(long)r * F + j];
        s += v; s2 += v * v;
    }
    atomicAdd(&csum[j], s);
    atomicAdd(&csq[j], s2);
}

__global__ __launch_bounds__(128)
void bn_finalize_kernel(const float* __restrict__ csum, const float* __restrict__ csq,
                        const float* __restrict__ gamma, const float* __restrict__ beta,
                        float* __restrict__ ss) {
    int j = threadIdx.x;
    float mean = csum[j] * (1.0f / NN);
    float var  = csq[j]  * (1.0f / NN) - mean * mean;
    float sc   = gamma[j] * rsqrtf(var + BN_EPS);
    ss[j]     = sc;
    ss[F + j] = beta[j] - mean * sc;
}

__global__ void bn_relu_kernel(float* __restrict__ Y, const float* __restrict__ ss) {
    long i = (long)blockIdx.x * blockDim.x + threadIdx.x;
    long total = (long)NN * F;
    long st = (long)gridDim.x * blockDim.x;
    for (; i < total; i += st) {
        int j = (int)(i % F);
        float v = Y[i] * ss[j] + ss[F + j];
        Y[i] = v > 0.0f ? v : 0.0f;
    }
}

// score[n] = agg[n].Wp_rel + x[n].Wp_root + bp  (wave per node, shuffle reduce)
__global__ __launch_bounds__(256)
void score_kernel(const float* __restrict__ agg, const float* __restrict__ X,
                  const float* __restrict__ Wpr, const float* __restrict__ Wpo,
                  const float* __restrict__ bp, float* __restrict__ score) {
    int gw = (int)(((long)blockIdx.x * blockDim.x + threadIdx.x) >> 5);
    int lane = threadIdx.x & 31;
    if (gw >= NN) return;
    float4 av = ((const float4*)(agg + (long)gw * F))[lane];
    float4 xv = ((const float4*)(X   + (long)gw * F))[lane];
    float4 rv = ((const float4*)Wpr)[lane];
    float4 ov = ((const float4*)Wpo)[lane];
    float s = av.x * rv.x + av.y * rv.y + av.z * rv.z + av.w * rv.w
            + xv.x * ov.x + xv.y * ov.y + xv.z * ov.z + xv.w * ov.w;
#pragma unroll
    for (int off = 16; off; off >>= 1) s += __shfl_xor(s, off, 32);
    if (lane == 0) score[gw] = s + bp[0];
}

// K = 1: per-graph argmax + tanh gate
__global__ __launch_bounds__(256)
void argmax_kernel(const float* __restrict__ score, int* __restrict__ perm,
                   float* __restrict__ gate) {
    __shared__ float sv[256];
    __shared__ int   si[256];
    int g = blockIdx.x, t = threadIdx.x;
    float best = -1e30f; int bi = g * NPG;
    for (int r = t; r < NPG; r += 256) {
        float v = score[g * NPG + r];
        if (v > best) { best = v; bi = g * NPG + r; }
    }
    sv[t] = best; si[t] = bi;
    __syncthreads();
    for (int s = 128; s; s >>= 1) {
        if (t < s && sv[t + s] > sv[t]) { sv[t] = sv[t + s]; si[t] = si[t + s]; }
        __syncthreads();
    }
    if (t == 0) { perm[g] = si[0]; gate[g] = tanhf(sv[0]); }
}

// out[g,o] = gate[g] * dot(x[perm[g]], Wm[:,o]) + bm[o]   (16x20, trivial)
__global__ __launch_bounds__(320)
void final_kernel(const float* __restrict__ X, const int* __restrict__ perm,
                  const float* __restrict__ gate, const float* __restrict__ Wm,
                  const float* __restrict__ bm, float* __restrict__ out) {
    int t = threadIdx.x;
    if (t >= NG * OUTC) return;
    int g = t / OUTC, o = t % OUTC;
    const float* xr = X + (long)perm[g] * F;
    float s = 0.0f;
#pragma unroll 4
    for (int k = 0; k < F; ++k) s += xr[k] * Wm[k * OUTC + o];
    out[t] = s * gate[g] + bm[o];
}

// ---------------- host launch ----------------
extern "C" void kernel_launch(void* const* d_in, const int* in_sizes, int n_in,
                              void* d_out, int out_size, void* d_ws, size_t ws_size,
                              hipStream_t stream) {
    (void)in_sizes; (void)n_in; (void)out_size; (void)ws_size;

    const float*     x_in = (const float*)d_in[0];
    const long long* ei   = (const long long*)d_in[1];   // int64 [2,NE]
    // d_in[2] = batch (derivable arithmetically; unused)
    // layer i params at 3 + 7*i: Wvn, bvn, Wrel, Wroot, b, gamma, beta
    const float* Wp_rel  = (const float*)d_in[24];
    const float* Wp_root = (const float*)d_in[25];
    const float* bp      = (const float*)d_in[26];
    const float* Wm      = (const float*)d_in[27];
    const float* bm      = (const float*)d_in[28];

    float* ws  = (float*)d_ws;
    float* B0  = ws;                         // y / next-layer x (aliased, 51.2 MB)
    float* B1  = B0 + (size_t)NN * F;        // x + vx broadcast
    float* B2  = B1 + (size_t)NN * F;        // agg (scatter target)
    float* vsum = B2 + (size_t)NN * F;       // 16*128
    float* csum = vsum + NG * F;             // 128   (contiguous with vsum for zeroing)
    float* csq  = csum + F;                  // 128
    float* vx   = csq + F;                   // 16*128
    float* ss   = vx + NG * F;               // 256 scale/shift
    float* score = ss + 2 * F;               // NN
    float* gate  = score + NN;               // 16
    int*   perm  = (int*)(gate + NG);        // 16

    // packed f16 weight fragments, 64B-aligned (v16h loads need 32B alignment)
    uintptr_t wp = ((uintptr_t)(perm + NG) + 63) & ~(uintptr_t)63;
    _Float16* W16a = (_Float16*)wp;          // 64 frags * 512 halves = 64KB each
    _Float16* W16b = W16a + 64 * 512;
    _Float16* W16c = W16b + 64 * 512;

    const long nf = (long)NN * F;
    const int gridNF = 2048;
    const int gemmBlocks = (NN + 127) / 128;       // 782
    const int scatBlocks = NE / 8;                 // 8 waves (edges) per 256-thr block
    const int scoreBlocks = (NN * 32 + 255) / 256; // wave per node

    const float* xcur = x_in;
    for (int L = 0; L < 3; ++L) {
        const float* Wvn   = (const float*)d_in[3 + 7 * L + 0];
        const float* bvn   = (const float*)d_in[3 + 7 * L + 1];
        const float* Wrel  = (const float*)d_in[3 + 7 * L + 2];
        const float* Wroot = (const float*)d_in[3 + 7 * L + 3];
        const float* bb    = (const float*)d_in[3 + 7 * L + 4];
        const float* gamma = (const float*)d_in[3 + 7 * L + 5];
        const float* beta  = (const float*)d_in[3 + 7 * L + 6];

        zero_kernel<<<gridNF, 256, 0, stream>>>(B2, nf);                  // agg = 0
        zero_kernel<<<4, 256, 0, stream>>>(vsum, (long)(NG * F + 2 * F)); // vsum+stats = 0
        pack_w_kernel<<<8, 256, 0, stream>>>(Wvn,   W16a);
        pack_w_kernel<<<8, 256, 0, stream>>>(Wrel,  W16b);
        pack_w_kernel<<<8, 256, 0, stream>>>(Wroot, W16c);

        vsum_kernel<<<NG, 128, 0, stream>>>(xcur, vsum);
        vn_mlp_kernel<<<1, 256, 0, stream>>>(vsum, W16a, bvn, vx);
        bcast_add_kernel<<<gridNF, 256, 0, stream>>>(xcur, vx, B1);
        scatter_add_kernel<<<scatBlocks, 256, 0, stream>>>(B1, ei, B2);
        gemm_dual_kernel<<<gemmBlocks, 256, 0, stream>>>(B2, B1, W16b, W16c, bb, B0, NN);
        colstats_kernel<<<512, 128, 0, stream>>>(B0, csum, csq);
        bn_finalize_kernel<<<1, 128, 0, stream>>>(csum, csq, gamma, beta, ss);
        bn_relu_kernel<<<gridNF, 256, 0, stream>>>(B0, ss);
        xcur = B0;
    }

    // SAG pooling: one more scatter, per-node score, per-graph argmax, tiny readout
    zero_kernel<<<gridNF, 256, 0, stream>>>(B2, nf);
    scatter_add_kernel<<<scatBlocks, 256, 0, stream>>>(B0, ei, B2);
    score_kernel<<<scoreBlocks, 256, 0, stream>>>(B2, B0, Wp_rel, Wp_root, bp, score);
    argmax_kernel<<<NG, 256, 0, stream>>>(score, perm, gate);
    final_kernel<<<1, 320, 0, stream>>>(B0, perm, gate, Wm, bm, (float*)d_out);
}
